// ViTInferenceNet_53730040873379
// MI455X (gfx1250) — compile-verified, matrix operand order checked by reading, not run on hardware
//
#include <hip/hip_runtime.h>
#include <cstdint>

#define S_LEN 512
#define NLAYER 4

typedef __bf16 bf16;
typedef bf16 bf16x4 __attribute__((ext_vector_type(4)));
typedef bf16 bf16x8 __attribute__((ext_vector_type(8)));
typedef bf16 bf16x16 __attribute__((ext_vector_type(16)));
typedef float v8f __attribute__((ext_vector_type(8)));
typedef unsigned int u32x4 __attribute__((ext_vector_type(4)));
typedef int i32x4 __attribute__((ext_vector_type(4)));
typedef int i32x8 __attribute__((ext_vector_type(8)));

// LDS layout: xs must be first (TDM descriptor uses lds_addr = 0).
struct SMem {
  float xs[S_LEN * 16];                       // 32 KB residual stream (f32)
  union { bf16 hb[S_LEN * 16];                // LN1 output (QKV phase input)
          bf16 ob[S_LEN * 16]; };             // attention output (out-proj input)
  bf16 qb[S_LEN * 16];                        // scaled Q, row-major
  bf16 kb[S_LEN * 16];                        // K, row-major
  bf16 vT[16 * S_LEN];                        // V transposed: [dim][token]
};                                            // total 96 KB

// Build a 16x32 bf16 A-operand from 8 contiguous bf16 (K=0..15 real, 16..31 zero pad).
// A layout: lane (half,m): element j<8 -> K = 8*half + j ; j>=8 -> K = 16 + 8*half + (j-8).
__device__ __forceinline__ bf16x16 load_a8(const bf16* p) {
  bf16x8 lo = *(const bf16x8*)p;
  bf16x16 r;
#pragma unroll
  for (int j = 0; j < 8; ++j) { r[j] = lo[j]; r[j + 8] = (bf16)0.0f; }
  return r;
}

__device__ __forceinline__ v8f wmma_bf16(bf16x16 a, bf16x16 b, v8f c) {
  return __builtin_amdgcn_wmma_f32_16x16x32_bf16(false, a, false, b, (short)0, c,
                                                 false, false);
}

// SWAPX16 across the two 16-lane halves of the wave (group-of-32 swizzle, xor=0x10).
__device__ __forceinline__ float swz16(float x) {
  return __builtin_bit_cast(
      float, __builtin_amdgcn_ds_swizzle(__builtin_bit_cast(int, x), 0x401f));
}

__global__ __launch_bounds__(256) void vit_fused_kernel(
    const float* __restrict__ x, const float* __restrict__ ln1_g,
    const float* __restrict__ ln1_b, const float* __restrict__ in_w,
    const float* __restrict__ in_b, const float* __restrict__ out_w,
    const float* __restrict__ out_b, const float* __restrict__ ln2_g,
    const float* __restrict__ ln2_b, const float* __restrict__ w1,
    const float* __restrict__ b1, const float* __restrict__ w2,
    const float* __restrict__ b2, const float* __restrict__ head_g,
    const float* __restrict__ head_b, const float* __restrict__ head_w,
    const float* __restrict__ head_bias, float* __restrict__ out) {
  __shared__ SMem sm;
  const int tid = threadIdx.x;
  const int lane = tid & 31;
  const int wv = tid >> 5;        // 8 waves
  const int half = lane >> 4;     // lane half (K-slot split)
  const int n16 = lane & 15;      // N/M index within 16
  const int b = blockIdx.x;

  // ---- Phase 0: TDM async load of x[b] (512x16 f32 = one 8192-elem row) into LDS ----
  {
    const float* xp = x + (size_t)b * S_LEN * 16;
    unsigned long long ga = (unsigned long long)(size_t)xp;
    // D# group0: count=1, lds_addr=0, global_addr, type=2
    u32x4 g0 = {1u, 0u, (unsigned)(ga & 0xffffffffull),
                (unsigned)((ga >> 32) & 0x1ffffffull) | 0x80000000u};
    // D# group1: data_size=4B(code 2); tensor_dim0=8192; tensor_dim1=1;
    //            tile_dim0=8192; tile_dim1=1; tensor_dim0_stride=8192
    i32x8 g1 = {0x00020000, 0x20000000, 0x00010000, 0x20000000, 1, 8192, 0, 0};
    i32x4 z4 = {0, 0, 0, 0};
    if (wv == 0) {
#if __clang_major__ >= 23
      i32x8 z8 = {0, 0, 0, 0, 0, 0, 0, 0};
      __builtin_amdgcn_tensor_load_to_lds(g0, g1, z4, z4, z8, 0);
#else
      __builtin_amdgcn_tensor_load_to_lds(g0, g1, z4, z4, 0);
#endif
      __builtin_amdgcn_s_wait_tensorcnt(0);
    }
  }
  __syncthreads();

  for (int l = 0; l < NLAYER; ++l) {
    // ---- LN1: xs -> hb (bf16), 2 rows per thread ----
    {
      const float* g = ln1_g + l * 16;
      const float* bb = ln1_b + l * 16;
      for (int r = tid; r < S_LEN; r += 256) {
        float v[16];
        float mu = 0.f;
#pragma unroll
        for (int d = 0; d < 16; ++d) { v[d] = sm.xs[r * 16 + d]; mu += v[d]; }
        mu *= (1.f / 16.f);
        float var = 0.f;
#pragma unroll
        for (int d = 0; d < 16; ++d) { float t = v[d] - mu; var += t * t; }
        var *= (1.f / 16.f);
        float rs = rsqrtf(var + 1e-5f);
#pragma unroll
        for (int d = 0; d < 16; ++d)
          sm.hb[r * 16 + d] = (bf16)((v[d] - mu) * rs * g[d] + bb[d]);
      }
    }
    __syncthreads();

    // ---- QKV projection: qkv = hb @ in_w^T + in_b  (WMMA, K=16 padded to 32) ----
    {
      const float* iw = in_w + l * 48 * 16;
      const float* ib = in_b + l * 48;
      // B layout: lane (half,n) element j<8 -> K = d = 8*half + j. Scale folded into Q.
      bf16x16 Bq, Bk, Bv;
#pragma unroll
      for (int j = 0; j < 8; ++j) {
        Bq[j] = (bf16)(iw[n16 * 16 + 8 * half + j] * 0.5f);
        Bk[j] = (bf16)(iw[(16 + n16) * 16 + 8 * half + j]);
        Bv[j] = (bf16)(iw[(32 + n16) * 16 + 8 * half + j]);
        Bq[j + 8] = (bf16)0.0f; Bk[j + 8] = (bf16)0.0f; Bv[j + 8] = (bf16)0.0f;
      }
      float bq = ib[n16] * 0.5f, bk = ib[16 + n16], bv = ib[32 + n16];
      for (int t = wv; t < 32; t += 8) {
        bf16x16 A = load_a8(&sm.hb[(t * 16 + n16) * 16 + 8 * half]);
        v8f z = {};
        v8f Dq = wmma_bf16(A, Bq, z);
        v8f Dk = wmma_bf16(A, Bk, z);
        v8f Dv = wmma_bf16(A, Bv, z);
        bf16x8 pv;
#pragma unroll
        for (int r = 0; r < 8; ++r) {
          int row = t * 16 + 8 * half + r;   // D layout: vgpr r -> M = r + 8*half
          sm.qb[row * 16 + n16] = (bf16)(Dq[r] + bq);
          sm.kb[row * 16 + n16] = (bf16)(Dk[r] + bk);
          pv[r] = (bf16)(Dv[r] + bv);
        }
        *(bf16x8*)&sm.vT[n16 * S_LEN + t * 16 + 8 * half] = pv;  // contiguous tokens
      }
    }
    __syncthreads();

    // ---- Attention: per (query-tile, head) unit; S^T = K·Q^T; online softmax; o^T = V^T·P^T ----
    for (int u = wv; u < 128; u += 8) {
      int qt = u >> 2, h = u & 3;
      // Q^T as B-operand, only head h's 4 dims nonzero (K slot d = 4h..4h+3)
      bf16x16 qB = {};
      if (half == (h >> 1)) {
        const bf16* qp = &sm.qb[(qt * 16 + n16) * 16 + 4 * h];
        if (h & 1) { qB[4] = qp[0]; qB[5] = qp[1]; qB[6] = qp[2]; qB[7] = qp[3]; }
        else       { qB[0] = qp[0]; qB[1] = qp[1]; qB[2] = qp[2]; qB[3] = qp[3]; }
      }
      float mrun = -3.4e38f, lsum = 0.f;
      v8f o = {};
      for (int kc = 0; kc < 16; ++kc) {   // 32 keys per step
        int k0 = kc * 32;
        bf16x16 A0 = load_a8(&sm.kb[(k0 + n16) * 16 + 8 * half]);
        bf16x16 A1 = load_a8(&sm.kb[(k0 + 16 + n16) * 16 + 8 * half]);
        v8f z = {};
        v8f S0 = wmma_bf16(A0, qB, z);    // S^T tile, keys k0..k0+15
        v8f S1 = wmma_bf16(A1, qB, z);    // keys k0+16..k0+31
        float tm = S0[0];
#pragma unroll
        for (int r = 1; r < 8; ++r) tm = fmaxf(tm, S0[r]);
#pragma unroll
        for (int r = 0; r < 8; ++r) tm = fmaxf(tm, S1[r]);
        tm = fmaxf(tm, swz16(tm));        // combine lane halves (per-query max)
        float mn = fmaxf(mrun, tm);
        float alpha = __expf(mrun - mn);
        mrun = mn;
        // P = exp(S^T - m): D layout of S^T is exactly the B-operand layout for PV.
        bf16x16 P;
        float ps = 0.f;
#pragma unroll
        for (int r = 0; r < 8; ++r) { float e = __expf(S0[r] - mn); ps += e; P[r] = (bf16)e; }
#pragma unroll
        for (int r = 0; r < 8; ++r) { float e = __expf(S1[r] - mn); ps += e; P[8 + r] = (bf16)e; }
        lsum = lsum * alpha + ps + swz16(ps);
        // V^T as A-operand: rows = all 16 dims, K = 32 keys (contiguous in vT)
        bf16x16 VA;
        {
          bf16x8 va0 = *(const bf16x8*)&sm.vT[n16 * S_LEN + k0 + 8 * half];
          bf16x8 va1 = *(const bf16x8*)&sm.vT[n16 * S_LEN + k0 + 16 + 8 * half];
#pragma unroll
          for (int j = 0; j < 8; ++j) { VA[j] = va0[j]; VA[8 + j] = va1[j]; }
        }
#pragma unroll
        for (int r = 0; r < 8; ++r) o[r] *= alpha;  // rescale accumulator
        o = wmma_bf16(VA, P, o);                    // o^T += V^T · P^T
      }
      float inv = __builtin_amdgcn_rcpf(lsum);
      // Write only head h's 4 output dims (rows d = 4h..4h+3 of o^T)
      if (half == (h >> 1)) {
        bf16x4 o4;
        if (h & 1) { o4[0] = (bf16)(o[4] * inv); o4[1] = (bf16)(o[5] * inv);
                     o4[2] = (bf16)(o[6] * inv); o4[3] = (bf16)(o[7] * inv); }
        else       { o4[0] = (bf16)(o[0] * inv); o4[1] = (bf16)(o[1] * inv);
                     o4[2] = (bf16)(o[2] * inv); o4[3] = (bf16)(o[3] * inv); }
        *(bf16x4*)&sm.ob[(qt * 16 + n16) * 16 + 4 * h] = o4;
      }
    }
    __syncthreads();

    // ---- Output projection + residual: xs += ob @ out_w^T + out_b (WMMA) ----
    {
      const float* ow = out_w + l * 16 * 16;
      float obias = out_b[l * 16 + n16];
      bf16x16 Bo;
#pragma unroll
      for (int j = 0; j < 8; ++j) {
        Bo[j] = (bf16)ow[n16 * 16 + 8 * half + j];
        Bo[j + 8] = (bf16)0.0f;
      }
      for (int t = wv; t < 32; t += 8) {
        bf16x16 A = load_a8(&sm.ob[(t * 16 + n16) * 16 + 8 * half]);
        v8f z = {};
        v8f D = wmma_bf16(A, Bo, z);
#pragma unroll
        for (int r = 0; r < 8; ++r) {
          int row = t * 16 + 8 * half + r;
          sm.xs[row * 16 + n16] += D[r] + obias;
        }
      }
    }
    __syncthreads();

    // ---- LN2 + MLP (16 -> 4 -> 16) + residual: pure VALU, 2 rows/thread ----
    {
      const float* g = ln2_g + l * 16;
      const float* bb = ln2_b + l * 16;
      const float* w1p = w1 + l * 4 * 16;
      const float* b1p = b1 + l * 4;
      const float* w2p = w2 + l * 16 * 4;
      const float* b2p = b2 + l * 16;
      for (int r = tid; r < S_LEN; r += 256) {
        float v[16];
        float mu = 0.f;
#pragma unroll
        for (int d = 0; d < 16; ++d) { v[d] = sm.xs[r * 16 + d]; mu += v[d]; }
        mu *= (1.f / 16.f);
        float var = 0.f;
#pragma unroll
        for (int d = 0; d < 16; ++d) { float t2 = v[d] - mu; var += t2 * t2; }
        var *= (1.f / 16.f);
        float rs = rsqrtf(var + 1e-5f);
        float nrm[16];
#pragma unroll
        for (int d = 0; d < 16; ++d) nrm[d] = (v[d] - mu) * rs * g[d] + bb[d];
        float h1[4];
#pragma unroll
        for (int j = 0; j < 4; ++j) {
          float s = b1p[j];
#pragma unroll
          for (int d = 0; d < 16; ++d) s += nrm[d] * w1p[j * 16 + d];
          h1[j] = 0.5f * s * (1.f + erff(s * 0.70710678f));  // exact gelu
        }
#pragma unroll
        for (int d = 0; d < 16; ++d) {
          float s = b2p[d];
#pragma unroll
          for (int j = 0; j < 4; ++j) s += h1[j] * w2p[d * 4 + j];
          sm.xs[r * 16 + d] = v[d] + s;
        }
      }
    }
    __syncthreads();
  }

  // ---- Head on CLS token: LN + (16 -> 16) linear; one output per lane ----
  if (tid < 16) {
    float v[16];
    float mu = 0.f;
#pragma unroll
    for (int d = 0; d < 16; ++d) { v[d] = sm.xs[d]; mu += v[d]; }
    mu *= (1.f / 16.f);
    float var = 0.f;
#pragma unroll
    for (int d = 0; d < 16; ++d) { float t2 = v[d] - mu; var += t2 * t2; }
    var *= (1.f / 16.f);
    float rs = rsqrtf(var + 1e-5f);
    float s = head_bias[tid];
#pragma unroll
    for (int d = 0; d < 16; ++d)
      s += ((v[d] - mu) * rs * head_g[d] + head_b[d]) * head_w[tid * 16 + d];
    out[b * 16 + tid] = s;
  }
}

extern "C" void kernel_launch(void* const* d_in, const int* in_sizes, int n_in,
                              void* d_out, int out_size, void* d_ws, size_t ws_size,
                              hipStream_t stream) {
  (void)in_sizes; (void)n_in; (void)out_size; (void)d_ws; (void)ws_size;
  vit_fused_kernel<<<256, 256, 0, stream>>>(
      (const float*)d_in[0], (const float*)d_in[1], (const float*)d_in[2],
      (const float*)d_in[3], (const float*)d_in[4], (const float*)d_in[5],
      (const float*)d_in[6], (const float*)d_in[7], (const float*)d_in[8],
      (const float*)d_in[9], (const float*)d_in[10], (const float*)d_in[11],
      (const float*)d_in[12], (const float*)d_in[13], (const float*)d_in[14],
      (const float*)d_in[15], (const float*)d_in[16], (float*)d_out);
}
